// SDF_TCNN_65996467470560
// MI455X (gfx1250) — compile-verified, matrix operand order checked by reading, not run on hardware
//
#include <hip/hip_runtime.h>
#include <hip/hip_bf16.h>
#include <stdint.h>

typedef __attribute__((ext_vector_type(16))) _Float16 v16h;
typedef __attribute__((ext_vector_type(8)))  float    v8f;

#define NLVL 16
#define TSIZE (1u << 19)

// Reassemble a 16x32-f16 WMMA A-fragment row from LDS (row-major f16, viewed as uints).
// ISA layout: lanes 0-15 hold K={0..7,16..23}, lanes 16-31 hold K={8..15,24..31};
// VGPR v holds a consecutive K pair -> uint indices {ubase..ubase+3} and {ubase+8..ubase+11}.
__device__ __forceinline__ v16h ld_fragA(const uint32_t* row, int ubase) {
  union { uint32_t u[8]; v16h h; } t;
#pragma unroll
  for (int i = 0; i < 4; ++i) {
    t.u[i]     = row[ubase + i];
    t.u[4 + i] = row[ubase + 8 + i];
  }
  return t.h;
}

__launch_bounds__(256, 1)
__global__ void sdf_tcnn_fused(const float* __restrict__ x,
                               const float* __restrict__ table,
                               const float* __restrict__ W1, const float* __restrict__ b1,
                               const float* __restrict__ W2, const float* __restrict__ b2,
                               const float* __restrict__ W3, const float* __restrict__ b3,
                               float* __restrict__ out, int npts) {
  // per-wave tiles: enc = 32 pts x 32 f16 feats (as 16 uints), h = 32 pts x 64 f16
  __shared__ uint32_t encS[8][32][16];   // 16 KB
  __shared__ _Float16 hS[8][32][64];     // 32 KB

  const int lane  = threadIdx.x & 31;
  const int wave  = threadIdx.x >> 5;
  const int n0    = lane & 15;
  const int hi    = lane >> 4;       // which 16-lane half
  const int ubOff = hi * 4;          // uint offset inside fragment row
  const int kBase = hi * 8;          // K base for B fragments / M base for C rows

  // ---------------- hash-grid encoding: one point per lane ----------------
  int p = blockIdx.x * 256 + wave * 32 + lane;
  if (p >= npts) p = npts - 1;       // clamp (no divergence before WMMA)
  const float px = (x[3 * p + 0] + 1.0f) * 0.5f;
  const float py = (x[3 * p + 1] + 1.0f) * 0.5f;
  const float pz = (x[3 * p + 2] + 1.0f) * 0.5f;

  float sgrow = 16.0f;               // BASE_RES * 1.3819^lvl
#pragma unroll
  for (int lvl = 0; lvl < NLVL; ++lvl) {
    const float scale = sgrow - 1.0f;
    sgrow *= 1.3819f;
    const int res = (int)ceilf(scale) + 1;
    const bool dense = ((long long)res * res * res) <= (long long)TSIZE;

    const float fx = px * scale + 0.5f;
    const float fy = py * scale + 0.5f;
    const float fz = pz * scale + 0.5f;
    const float gx = floorf(fx), gy = floorf(fy), gz = floorf(fz);
    const float wx = fx - gx, wy = fy - gy, wz = fz - gz;
    const int ix = (int)gx, iy = (int)gy, iz = (int)gz;

    const float* tl = table + (size_t)lvl * TSIZE * 2u;
    float a0 = 0.0f, a1 = 0.0f;
#pragma unroll
    for (int c = 0; c < 8; ++c) {
      const int ox = (c >> 2) & 1, oy = (c >> 1) & 1, oz = c & 1; // meshgrid-ij order
      const uint32_t cx = (uint32_t)(ix + ox);
      const uint32_t cy = (uint32_t)(iy + oy);
      const uint32_t cz = (uint32_t)(iz + oz);
      uint32_t idx;
      if (dense) {
        idx = cx + cy * (uint32_t)res + cz * (uint32_t)(res * res);
      } else {
        idx = (cx * 1u) ^ (cy * 2654435761u) ^ (cz * 805459861u);
        idx &= (TSIZE - 1u);
      }
      const float2 f = *reinterpret_cast<const float2*>(tl + 2u * (size_t)idx); // L2-resident gather
      const float wgt = (ox ? wx : 1.0f - wx) * (oy ? wy : 1.0f - wy) * (oz ? wz : 1.0f - wz);
      a0 = fmaf(wgt, f.x, a0);
      a1 = fmaf(wgt, f.y, a1);
    }
    union { _Float16 h[2]; uint32_t u; } pk;
    pk.h[0] = (_Float16)a0;
    pk.h[1] = (_Float16)a1;
    encS[wave][lane][lvl] = pk.u;    // f16 cols {2*lvl, 2*lvl+1}
  }

  // ---------------- preload weight B-fragments into registers ----------------
  // B layout mirrors A: lane (half hi) holds col N=n0, K = kBase + {0..7, 16..23}
  v16h B1[4];
#pragma unroll
  for (int ct = 0; ct < 4; ++ct) {
    const int col = 16 * ct + n0;
#pragma unroll
    for (int e = 0; e < 16; ++e) {
      const int K = kBase + (e < 8 ? e : e + 8);
      B1[ct][e] = (_Float16)W1[K * 64 + col];
    }
  }
  v16h B2[4][2];
#pragma unroll
  for (int ct = 0; ct < 4; ++ct) {
    const int col = 16 * ct + n0;
#pragma unroll
    for (int kc = 0; kc < 2; ++kc) {
#pragma unroll
      for (int e = 0; e < 16; ++e) {
        const int K = kc * 32 + kBase + (e < 8 ? e : e + 8);
        B2[ct][kc][e] = (_Float16)W2[K * 64 + col];
      }
    }
  }
  float b1v[4], b2v[4], w3v[4];
#pragma unroll
  for (int ct = 0; ct < 4; ++ct) {
    b1v[ct] = b1[16 * ct + n0];
    b2v[ct] = b2[16 * ct + n0];
    w3v[ct] = W3[16 * ct + n0];
  }
  const float bias3 = b3[0];

  // ---------------- MLP: two 16-point subtiles per wave ----------------
#pragma unroll
  for (int st = 0; st < 2; ++st) {
    const int row = st * 16 + n0;

    // Layer 1: enc(16x32) @ W1(32x64), 4 column tiles, one shared A fragment
    const v16h a1 = ld_fragA(&encS[wave][row][0], ubOff);
#pragma unroll
    for (int ct = 0; ct < 4; ++ct) {
      v8f c = {};
      c = __builtin_amdgcn_wmma_f32_16x16x32_f16(false, a1, false, B1[ct],
                                                 (short)0, c, false, false);
#pragma unroll
      for (int r = 0; r < 8; ++r) {                 // C row M = r + kBase
        const float v = fmaxf(c[r] + b1v[ct], 0.0f);
        hS[wave][st * 16 + r + kBase][16 * ct + n0] = (_Float16)v;
      }
    }

    // Layer 2 A fragments: loop-invariant across column tiles -> load once
    const uint32_t* hrow = reinterpret_cast<const uint32_t*>(&hS[wave][row][0]);
    v16h a2[2];
#pragma unroll
    for (int kc = 0; kc < 2; ++kc)
      a2[kc] = ld_fragA(hrow, kc * 16 + ubOff);

    // Layer 2: h(16x64) @ W2(64x64), fused with layer-3 dot against W3
    float part[8];
#pragma unroll
    for (int r = 0; r < 8; ++r) part[r] = 0.0f;
#pragma unroll
    for (int ct = 0; ct < 4; ++ct) {
      v8f c2 = {};
#pragma unroll
      for (int kc = 0; kc < 2; ++kc) {
        c2 = __builtin_amdgcn_wmma_f32_16x16x32_f16(false, a2[kc], false, B2[ct][kc],
                                                    (short)0, c2, false, false);
      }
#pragma unroll
      for (int r = 0; r < 8; ++r) {
        const float v = fmaxf(c2[r] + b2v[ct], 0.0f);
        part[r] = fmaf(v, w3v[ct], part[r]);
      }
    }

    // Layer 3 reduce: sum over 16 columns held across each 16-lane half
#pragma unroll
    for (int r = 0; r < 8; ++r) {
#pragma unroll
      for (int m = 1; m <= 8; m <<= 1)
        part[r] += __shfl_xor(part[r], m, 32);
    }
    if (n0 == 0) {
      const int base = blockIdx.x * 256 + wave * 32 + st * 16 + kBase;
#pragma unroll
      for (int r = 0; r < 8; ++r) {
        const int q = base + r;
        if (q < npts) out[q] = part[r] + bias3;
      }
    }
  }
}

extern "C" void kernel_launch(void* const* d_in, const int* in_sizes, int n_in,
                              void* d_out, int out_size, void* d_ws, size_t ws_size,
                              hipStream_t stream) {
  const float* x  = (const float*)d_in[0];
  const float* tb = (const float*)d_in[1];
  const float* W1 = (const float*)d_in[2];
  const float* b1 = (const float*)d_in[3];
  const float* W2 = (const float*)d_in[4];
  const float* b2 = (const float*)d_in[5];
  const float* W3 = (const float*)d_in[6];
  const float* b3 = (const float*)d_in[7];
  float* out = (float*)d_out;

  const int npts   = in_sizes[0] / 3;       // 1048576
  const int blocks = (npts + 255) / 256;    // 256 pts per block (8 waves x 32)
  sdf_tcnn_fused<<<blocks, 256, 0, stream>>>(x, tb, W1, b1, W2, b2, W3, b3, out, npts);
}
